// RNN_87883620811027
// MI455X (gfx1250) — compile-verified, matrix operand order checked by reading, not run on hardware
//
#include <hip/hip_runtime.h>

#define DIMS 2048
#define SEQ  8192

typedef __attribute__((ext_vector_type(16))) _Float16 v16h;
typedef __attribute__((ext_vector_type(8)))  _Float16 v8h;
typedef __attribute__((ext_vector_type(4)))  _Float16 v4h;
typedef __attribute__((ext_vector_type(8)))  float    v8f;

// ---------------------------------------------------------------------------
// f32 -> f16 conversion (vectorized 4-wide; all tensor sizes divisible by 4)
// ---------------------------------------------------------------------------
__global__ __launch_bounds__(256) void cvt_f32_to_f16(
    const float4* __restrict__ in, v4h* __restrict__ out, int n4) {
  int i = blockIdx.x * 256 + threadIdx.x;
  if (i < n4) {
    float4 f = in[i];
    v4h h;
    h[0] = (_Float16)f.x;
    h[1] = (_Float16)f.y;
    h[2] = (_Float16)f.z;
    h[3] = (_Float16)f.w;
    out[i] = h;
  }
}

// ---------------------------------------------------------------------------
// State init: h ping buffer = h0, reset grid-barrier counter (every call!)
// ---------------------------------------------------------------------------
__global__ __launch_bounds__(256) void init_state(
    const float* __restrict__ h0, float* __restrict__ hbuf,
    unsigned* __restrict__ counter) {
  int i = blockIdx.x * 256 + threadIdx.x;
  if (i < DIMS) hbuf[i] = h0[i];
  if (i == 0) *counter = 0u;
}

// ---------------------------------------------------------------------------
// Phase 1: xproj[s, d] = sum_k inputs[s,k] * W_hi[d,k] + b[d]
// WMMA f32_16x16x32_f16. Block = 8 waves; each wave: 2 M-tiles x 4 N-tiles.
// Block tile: 256 rows (M) x 64 cols (N). Grid: (DIMS/64, SEQ/256).
//
// Fragment loads (wave32):
//   A 16x32 f16 : lane holds row (lane&15), 16 contiguous K at k0+(lane>>4)*16
//   B 32x16 f16 : lane holds col (lane&15) = W_hi row, same contiguous K range
//   C 16x16 f32 : VGPR r -> row r + 8*(lane>>4), col lane&15
// ---------------------------------------------------------------------------
__global__ __launch_bounds__(256) void xproj_gemm(
    const _Float16* __restrict__ A, const _Float16* __restrict__ W,
    const float* __restrict__ bias, float* __restrict__ C) {
  const int lane = threadIdx.x & 31;
  const int wave = threadIdx.x >> 5;
  const int l16  = lane & 15;
  const int lhi  = lane >> 4;                 // K-half selector (0/1)
  const int nbase = blockIdx.x * 64;
  const int mbase = blockIdx.y * 256 + wave * 32;

  const _Float16* a0 = A + (size_t)(mbase + l16) * DIMS + lhi * 16;
  const _Float16* a1 = a0 + (size_t)16 * DIMS;
  const _Float16* w0 = W + (size_t)(nbase + l16) * DIMS + lhi * 16;

  v8f acc[2][4] = {};
  for (int k0 = 0; k0 < DIMS; k0 += 32) {
    v16h af0 = *(const v16h*)(a0 + k0);
    v16h af1 = *(const v16h*)(a1 + k0);
#pragma unroll
    for (int j = 0; j < 4; ++j) {
      v16h bf = *(const v16h*)(w0 + (size_t)(j * 16) * DIMS + k0);
      acc[0][j] = __builtin_amdgcn_wmma_f32_16x16x32_f16(
          false, af0, false, bf, (short)0, acc[0][j], false, false);
      acc[1][j] = __builtin_amdgcn_wmma_f32_16x16x32_f16(
          false, af1, false, bf, (short)0, acc[1][j], false, false);
    }
  }

#pragma unroll
  for (int j = 0; j < 4; ++j) {
    const int col = nbase + j * 16 + l16;
    const float bv = bias[col];
#pragma unroll
    for (int mi = 0; mi < 2; ++mi) {
      const int rbase = mbase + mi * 16 + lhi * 8;
#pragma unroll
      for (int r = 0; r < 8; ++r) {
        C[(size_t)(rbase + r) * DIMS + col] = acc[mi][j][r] + bv;
      }
    }
  }
}

// ---------------------------------------------------------------------------
// Phase 2: persistent recurrence kernel.
//   h_t = tanh(xproj[t] + W_hh @ h_{t-1});  out[t] = x[t] + h_t
// 64 blocks x 256 threads. Block owns 32 rows; 8 threads per row, each thread
// a contiguous 256-col segment. h ping-pongs between two global buffers;
// device-wide sync via atomic counter with monotonically increasing target.
// W_hh (f16, 8 MB) stays L2-resident; h_prev staged through LDS each step.
// ---------------------------------------------------------------------------
#define RNN_BLOCKS 64
#define ROWS_PER_BLOCK (DIMS / RNN_BLOCKS)   // 32
#define TPR (256 / ROWS_PER_BLOCK)           // 8 threads per row
#define SEG (DIMS / TPR)                     // 256 cols per thread

__global__ __launch_bounds__(256) void rnn_recur(
    const float* __restrict__ x, const float* __restrict__ xproj,
    const _Float16* __restrict__ Whh, float* __restrict__ hping,
    float* __restrict__ out, unsigned* __restrict__ counter) {
  __shared__ float hsh[DIMS];
  const int tid = threadIdx.x;
  const int rowLocal = tid / TPR;
  const int sub = tid % TPR;
  const int row = blockIdx.x * ROWS_PER_BLOCK + rowLocal;
  const _Float16* wrow = Whh + (size_t)row * DIMS + sub * SEG;
  float* hb0 = hping;
  float* hb1 = hping + DIMS;

  for (int t = 0; t < SEQ; ++t) {
    const float* hprev = (t & 1) ? hb1 : hb0;
    float* hnext       = (t & 1) ? hb0 : hb1;

    // stage h_{t-1} into LDS (coalesced)
    for (int i = tid; i < DIMS; i += 256) hsh[i] = hprev[i];
    __syncthreads();

    // partial dot product: 256 f16 weights -> f32 FMA
    float acc = 0.f;
    const float* hseg = hsh + sub * SEG;
#pragma unroll 4
    for (int i = 0; i < SEG; i += 8) {
      v8h wv = *(const v8h*)(wrow + i);
#pragma unroll
      for (int j = 0; j < 8; ++j) acc = fmaf((float)wv[j], hseg[i + j], acc);
    }
    // reduce across the 8 sub-lanes (contiguous within a wave32)
#pragma unroll
    for (int off = TPR >> 1; off > 0; off >>= 1)
      acc += __shfl_down(acc, off, TPR);

    if (sub == 0) {
      const size_t idx = (size_t)t * DIMS + row;
      float hv = tanhf(xproj[idx] + acc);
      hnext[row] = hv;
      out[idx] = x[idx] + hv;
    }

    // device-wide barrier (release/acquire)
    __syncthreads();
    if (tid == 0) {
      __threadfence();
      atomicAdd(counter, 1u);
      const unsigned target = (unsigned)(t + 1) * (unsigned)gridDim.x;
      while (__hip_atomic_load(counter, __ATOMIC_RELAXED,
                               __HIP_MEMORY_SCOPE_AGENT) < target) {
        __builtin_amdgcn_s_sleep(1);
      }
      __threadfence();
    }
    __syncthreads();
  }
}

// ---------------------------------------------------------------------------
// Launch. Inputs: [0]=inputs f32[SEQ,DIMS], [1]=h0 f32[DIMS],
//         [2]=W_hi f32[DIMS,DIMS], [3]=W_hh f32[DIMS,DIMS], [4]=b f32[DIMS]
// Workspace layout: A(f16) 32MB | W_hi(f16) 8MB | W_hh(f16) 8MB |
//                   xproj(f32) 64MB | h ping-pong 16KB | counter
// ---------------------------------------------------------------------------
extern "C" void kernel_launch(void* const* d_in, const int* in_sizes, int n_in,
                              void* d_out, int out_size, void* d_ws, size_t ws_size,
                              hipStream_t stream) {
  (void)in_sizes; (void)n_in; (void)out_size; (void)ws_size;
  const float* x   = (const float*)d_in[0];
  const float* h0  = (const float*)d_in[1];
  const float* Whi = (const float*)d_in[2];
  const float* Whh = (const float*)d_in[3];
  const float* b   = (const float*)d_in[4];
  float* out = (float*)d_out;

  char* ws = (char*)d_ws;
  _Float16* Ah   = (_Float16*)ws;                                   // SEQ*DIMS
  _Float16* Wih  = (_Float16*)(ws + (size_t)SEQ * DIMS * 2);        // DIMS*DIMS
  _Float16* Whh16 = Wih + (size_t)DIMS * DIMS;                      // DIMS*DIMS
  float* xproj = (float*)((char*)(Whh16 + (size_t)DIMS * DIMS));    // SEQ*DIMS
  float* hbuf  = xproj + (size_t)SEQ * DIMS;                        // 2*DIMS
  unsigned* counter = (unsigned*)(hbuf + 2 * DIMS);

  const int nA = SEQ * DIMS / 4, nW = DIMS * DIMS / 4;
  cvt_f32_to_f16<<<(nA + 255) / 256, 256, 0, stream>>>((const float4*)x,   (v4h*)Ah,    nA);
  cvt_f32_to_f16<<<(nW + 255) / 256, 256, 0, stream>>>((const float4*)Whi, (v4h*)Wih,   nW);
  cvt_f32_to_f16<<<(nW + 255) / 256, 256, 0, stream>>>((const float4*)Whh, (v4h*)Whh16, nW);
  init_state<<<(DIMS + 255) / 256, 256, 0, stream>>>(h0, hbuf, counter);

  dim3 gg(DIMS / 64, SEQ / 256);
  xproj_gemm<<<gg, 256, 0, stream>>>(Ah, Wih, b, xproj);

  rnn_recur<<<RNN_BLOCKS, 256, 0, stream>>>(x, xproj, Whh16, hbuf, out, counter);
}
